// PoseRNN_12455405158526
// MI455X (gfx1250) — matrix operand order repaired and checked by reading, not compiled, not run on hardware
//
#include <hip/hip_runtime.h>
#include <math.h>

typedef unsigned short u16;
typedef unsigned int   u32;
typedef __attribute__((ext_vector_type(16))) __bf16 v16bf;
typedef __attribute__((ext_vector_type(8)))  float  v8f;

#define LDS_STRIDE 40   // u16 units; multiple of 8 so 16B vector LDS ops stay aligned

__device__ __forceinline__ u16 f2b(float f) {
  u32 u = __float_as_uint(f);
  u32 r = u + 0x7FFFu + ((u >> 16) & 1u);   // round-to-nearest-even
  return (u16)(r >> 16);
}
__device__ __forceinline__ float b2f(u16 s) {
  return __uint_as_float(((u32)s) << 16);
}
__device__ __forceinline__ float sigmoidf_(float x) { return 1.f / (1.f + __expf(-x)); }
__device__ __forceinline__ float softplusf_(float x) { return x > 20.f ? x : log1pf(__expf(x)); }

union FragBF { v16bf v; u32 u[8]; };
union FragF  { v8f   v; float f[8]; };
union Pack8  { uint4 q; u16 s[8]; };

// Load a 16x32 bf16 A fragment from LDS (row-major, LDS_STRIDE pitch).
__device__ __forceinline__ void load_afrag(FragBF& fr, const u16* As, int rowBase, int lane) {
  const int row  = rowBase + (lane & 15);
  const int half = lane >> 4;
  const u16* p = &As[row * LDS_STRIDE];
#pragma unroll
  for (int j = 0; j < 4; ++j) fr.u[j]     = *(const u32*)(p + half * 8 + 2 * j);
#pragma unroll
  for (int j = 0; j < 4; ++j) fr.u[4 + j] = *(const u32*)(p + 16 + half * 8 + 2 * j);
}
// Load a 32x16 bf16 B fragment from LDS stored transposed as Bs[n][k].
__device__ __forceinline__ void load_bfrag(FragBF& fr, const u16* Bs, int colBase, int lane) {
  const int half = lane >> 4;
  const u16* p = &Bs[(colBase + (lane & 15)) * LDS_STRIDE];
#pragma unroll
  for (int j = 0; j < 4; ++j) fr.u[j]     = *(const u32*)(p + half * 8 + 2 * j);
#pragma unroll
  for (int j = 0; j < 4; ++j) fr.u[4 + j] = *(const u32*)(p + 16 + half * 8 + 2 * j);
}

// ---------------------------------------------------------------------------
// Generic bf16 GEMM: C[M,N] = A[M,K] * B[K,N]; A,B bf16 row-major.
// Block = 256 threads (8 waves). Block tile 64x128; wave tile 32x32
// (2 A frags x 2 B frags -> 4 independent WMMAs per K step).
// A tile is staged with GLOBAL_LOAD_ASYNC_TO_LDS_B128 (ASYNCcnt DMA path);
// B tile is staged through VGPRs because it is transposed into Bs[n][k].
// mode: 0 = fp32 store, 1 = bf16 store, 2 = bf16 store + leakyReLU(0.2)
// ---------------------------------------------------------------------------
__global__ __launch_bounds__(256) void wmma_gemm_bf16(
    const u16* __restrict__ A, int lda,
    const u16* __restrict__ Bw, int ldb,
    float* __restrict__ Cf, u16* __restrict__ Cb, int ldc,
    int Kp, int mode)
{
  __shared__ u16 As[64 * LDS_STRIDE];
  __shared__ u16 Bs[128 * LDS_STRIDE];   // transposed: Bs[n][k]

  const int tid  = threadIdx.x;
  const int lane = tid & 31;
  const int wave = tid >> 5;
  const int wm   = wave & 1;     // 2 M slabs of 32
  const int wn   = wave >> 1;    // 4 N slabs of 32
  const int m0   = blockIdx.y * 64;
  const int n0   = blockIdx.x * 128;

  FragF acc00, acc01, acc10, acc11;   // [A-sub][B-sub]
#pragma unroll
  for (int i = 0; i < 8; ++i) {
    acc00.f[i] = 0.f; acc01.f[i] = 0.f; acc10.f[i] = 0.f; acc11.f[i] = 0.f;
  }

  const int ar = tid >> 2;            // 0..63  A tile row
  const int ac = (tid & 3) * 8;       // A tile k-chunk
  const int br = tid >> 3;            // 0..31  B tile k row
  const int bc = (tid & 7) * 16;      // 16 cols per thread

  const unsigned long long aBase = (unsigned long long)A;
  const u32 ldsA  = (u32)(size_t)&As[ar * LDS_STRIDE + ac];     // per-lane LDS dest
  const u32 gOffA = (u32)(((size_t)(m0 + ar) * lda + ac) * sizeof(u16));
  const u16* bgBase = Bw + (size_t)br * ldb + n0 + bc;

  for (int kt = 0; kt < Kp; kt += 32) {
    Pack8 bq0, bq1;
    bq0.q = *(const uint4*)(bgBase + (size_t)kt * ldb);
    bq1.q = *(const uint4*)(bgBase + (size_t)kt * ldb + 8);
    if (kt + 32 < Kp)
      __builtin_prefetch(bgBase + (size_t)(kt + 32) * ldb, 0, 1);
    __syncthreads();   // previous iteration's fragment reads must finish first
    {
      // async DMA: 16 bytes/lane of the A tile, global -> LDS (ASYNCcnt)
      const u32 gOff = gOffA + (u32)(kt * sizeof(u16));
      asm volatile("global_load_async_to_lds_b128 %0, %1, %2"
                   :
                   : "v"(ldsA), "v"(gOff), "s"(aBase)
                   : "memory");
    }
#pragma unroll
    for (int i = 0; i < 8; ++i) Bs[(bc + i) * LDS_STRIDE + br]     = bq0.s[i];
#pragma unroll
    for (int i = 0; i < 8; ++i) Bs[(bc + 8 + i) * LDS_STRIDE + br] = bq1.s[i];
    asm volatile("s_wait_asynccnt 0x0" ::: "memory");
    __syncthreads();

    FragBF a0, a1, b0, b1;
    load_afrag(a0, As, wm * 32,      lane);
    load_afrag(a1, As, wm * 32 + 16, lane);
    load_bfrag(b0, Bs, wn * 32,      lane);
    load_bfrag(b1, Bs, wn * 32 + 16, lane);

    acc00.v = __builtin_amdgcn_wmma_f32_16x16x32_bf16(false, a0.v, false, b0.v, (short)0, acc00.v, false, false);
    acc01.v = __builtin_amdgcn_wmma_f32_16x16x32_bf16(false, a0.v, false, b1.v, (short)0, acc01.v, false, false);
    acc10.v = __builtin_amdgcn_wmma_f32_16x16x32_bf16(false, a1.v, false, b0.v, (short)0, acc10.v, false, false);
    acc11.v = __builtin_amdgcn_wmma_f32_16x16x32_bf16(false, a1.v, false, b1.v, (short)0, acc11.v, false, false);
  }

  const int half = lane >> 4;
  const int cA = n0 + wn * 32 + (lane & 15);
#pragma unroll
  for (int am = 0; am < 2; ++am) {
#pragma unroll
    for (int bn = 0; bn < 2; ++bn) {
      const FragF& a = am == 0 ? (bn == 0 ? acc00 : acc01) : (bn == 0 ? acc10 : acc11);
#pragma unroll
      for (int j = 0; j < 8; ++j) {
        const int row = m0 + wm * 32 + am * 16 + half * 8 + j;
        const size_t i0 = (size_t)row * ldc + cA + bn * 16;
        float v = a.f[j];
        if (mode == 0) Cf[i0] = v;
        else {
          if (mode == 2) v = v >= 0.f ? v : 0.2f * v;
          Cb[i0] = f2b(v);
        }
      }
    }
  }
}

// ---------------------------------------------------------------------------
// Implicit-GEMM conv (4x4 kernel): A rows are output pixels (NHWC gather from
// bf16 input), B is pre-transposed weight (Kp x Co, bf16).
// Block tile 128x64; wave tile 32x32 (4 WMMAs per K step).
// ---------------------------------------------------------------------------
__global__ __launch_bounds__(256) void wmma_conv_bf16(
    const u16* __restrict__ in, int Hin, int Win, int Cin,
    int Hout, int Wout, int stride, int pad, int Kdim,
    const u16* __restrict__ Bw, int Co,
    float* __restrict__ Cf, u16* __restrict__ Cb, int mode, int Kp)
{
  __shared__ u16 As[128 * LDS_STRIDE];
  __shared__ u16 Bs[64 * LDS_STRIDE];

  const int tid  = threadIdx.x;
  const int lane = tid & 31;
  const int wave = tid >> 5;
  const int wm   = wave & 3;     // 4 M slabs of 32
  const int wn   = wave >> 2;    // 2 N slabs of 32
  const int m0   = blockIdx.y * 128;
  const int n0   = blockIdx.x * 64;

  FragF acc00, acc01, acc10, acc11;
#pragma unroll
  for (int i = 0; i < 8; ++i) {
    acc00.f[i] = 0.f; acc01.f[i] = 0.f; acc10.f[i] = 0.f; acc11.f[i] = 0.f;
  }

  const int ar = tid >> 1;            // 0..127 A tile row
  const int ac = (tid & 1) * 16;      // 16 k per thread
  const int br = tid >> 3;            // 0..31  B tile k row
  const int bc = (tid & 7) * 8;       // 8 cols per thread

  // decompose this thread's A row (hoisted out of K loop)
  const int HW  = Hout * Wout;
  const int m   = m0 + ar;
  const int img = m / HW;
  const int rem = m - img * HW;
  const int oy  = rem / Wout;
  const int ox  = rem - oy * Wout;

  const u16* bgBase = Bw + (size_t)br * Co + n0 + bc;

  for (int kt = 0; kt < Kp; kt += 32) {
    u16 av[16];
#pragma unroll
    for (int i = 0; i < 16; ++i) {
      const int kk = kt + ac + i;
      u16 val = 0;
      if (kk < Kdim) {
        const int c  = kk >> 4;
        const int rr = kk & 15;
        const int ky = rr >> 2;
        const int kx = rr & 3;
        const int iy = oy * stride - pad + ky;
        const int ix = ox * stride - pad + kx;
        if (iy >= 0 && iy < Hin && ix >= 0 && ix < Win)
          val = in[(((size_t)img * Hin + iy) * Win + ix) * Cin + c];
      }
      av[i] = val;
    }
    Pack8 bq;
    bq.q = *(const uint4*)(bgBase + (size_t)kt * Co);
    __syncthreads();
    *(uint4*)&As[ar * LDS_STRIDE + ac]     = *(const uint4*)&av[0];
    *(uint4*)&As[ar * LDS_STRIDE + ac + 8] = *(const uint4*)&av[8];
#pragma unroll
    for (int i = 0; i < 8; ++i) Bs[(bc + i) * LDS_STRIDE + br] = bq.s[i];
    __syncthreads();

    FragBF a0, a1, b0, b1;
    load_afrag(a0, As, wm * 32,      lane);
    load_afrag(a1, As, wm * 32 + 16, lane);
    load_bfrag(b0, Bs, wn * 32,      lane);
    load_bfrag(b1, Bs, wn * 32 + 16, lane);

    acc00.v = __builtin_amdgcn_wmma_f32_16x16x32_bf16(false, a0.v, false, b0.v, (short)0, acc00.v, false, false);
    acc01.v = __builtin_amdgcn_wmma_f32_16x16x32_bf16(false, a0.v, false, b1.v, (short)0, acc01.v, false, false);
    acc10.v = __builtin_amdgcn_wmma_f32_16x16x32_bf16(false, a1.v, false, b0.v, (short)0, acc10.v, false, false);
    acc11.v = __builtin_amdgcn_wmma_f32_16x16x32_bf16(false, a1.v, false, b1.v, (short)0, acc11.v, false, false);
  }

  const int half = lane >> 4;
  const int cA = n0 + wn * 32 + (lane & 15);
#pragma unroll
  for (int am = 0; am < 2; ++am) {
#pragma unroll
    for (int bn = 0; bn < 2; ++bn) {
      const FragF& a = am == 0 ? (bn == 0 ? acc00 : acc01) : (bn == 0 ? acc10 : acc11);
#pragma unroll
      for (int j = 0; j < 8; ++j) {
        const int row = m0 + wm * 32 + am * 16 + half * 8 + j;
        const size_t i0 = (size_t)row * Co + cA + bn * 16;
        float v = a.f[j];
        if (mode == 0) Cf[i0] = v;
        else {
          if (mode == 2) v = v >= 0.f ? v : 0.2f * v;
          Cb[i0] = f2b(v);
        }
      }
    }
  }
}

// ---------------------------------------------------------------------------
// Aux kernels
// ---------------------------------------------------------------------------
__global__ void cvt_f2b_flat(const float* __restrict__ src, u16* __restrict__ dst, long n) {
  long i = (long)blockIdx.x * blockDim.x + threadIdx.x;
  if (i < n) dst[i] = f2b(src[i]);
}
__global__ void zero_u32_flat(u32* __restrict__ p, long n) {
  long i = (long)blockIdx.x * blockDim.x + threadIdx.x;
  if (i < n) p[i] = 0u;
}
__global__ void copy_b2b(const u16* __restrict__ src, int sld, u16* __restrict__ dst, int dld, int rows, int cols) {
  int i = blockIdx.x * blockDim.x + threadIdx.x;
  if (i >= rows * cols) return;
  int r = i / cols, c = i - r * cols;
  dst[(size_t)r * dld + c] = src[(size_t)r * sld + c];
}
__global__ void copy_f2b(const float* __restrict__ src, int sld, u16* __restrict__ dst, int dld, int rows, int cols) {
  int i = blockIdx.x * blockDim.x + threadIdx.x;
  if (i >= rows * cols) return;
  int r = i / cols, c = i - r * cols;
  dst[(size_t)r * dld + c] = f2b(src[(size_t)r * sld + c]);
}
__global__ void copy_f2f(const float* __restrict__ src, int sld, float* __restrict__ dst, int dld, int rows, int cols) {
  int i = blockIdx.x * blockDim.x + threadIdx.x;
  if (i >= rows * cols) return;
  int r = i / cols, c = i - r * cols;
  dst[(size_t)r * dld + c] = src[(size_t)r * sld + c];
}
// conv weight (Co,Cin,4,4) fp32 -> (Kp x Co) bf16, K index = c*16 + ky*4 + kx
__global__ void convw_t(const float* __restrict__ w, u16* __restrict__ wt, int Co, int Cin, int Kdim, int Kp) {
  int i = blockIdx.x * blockDim.x + threadIdx.x;
  if (i >= Kp * Co) return;
  int n = i % Co, k = i / Co;
  u16 v = 0;
  if (k < Kdim) {
    int c = k >> 4, r = k & 15;
    v = f2b(w[((size_t)n * Cin + c) * 16 + r]);
  }
  wt[(size_t)k * Co + n] = v;
}
// linear weight (N,K) fp32 -> wt[k*ldwt + n] bf16
__global__ void linw_t(const float* __restrict__ w, u16* __restrict__ wt, int N, int K, int ldwt) {
  long i = (long)blockIdx.x * blockDim.x + threadIdx.x;
  if (i >= (long)N * K) return;
  int n = (int)(i % N), k = (int)(i / N);
  wt[(size_t)k * ldwt + n] = f2b(w[(size_t)n * K + k]);
}
__global__ __launch_bounds__(256) void bn_stats(const u16* __restrict__ act, int M, int C,
                                                float* __restrict__ mean, float* __restrict__ var) {
  int c = blockIdx.x, tid = threadIdx.x;
  float s = 0.f, s2 = 0.f;
  for (int r = tid; r < M; r += 256) {
    float v = b2f(act[(size_t)r * C + c]);
    s += v; s2 += v * v;
  }
  __shared__ float sm[256], sm2[256];
  sm[tid] = s; sm2[tid] = s2; __syncthreads();
  for (int o = 128; o > 0; o >>= 1) {
    if (tid < o) { sm[tid] += sm[tid + o]; sm2[tid] += sm2[tid + o]; }
    __syncthreads();
  }
  if (tid == 0) { float mu = sm[0] / M; mean[c] = mu; var[c] = sm2[0] / M - mu * mu; }
}
__global__ void bn_apply_leaky(u16* __restrict__ act, const float* __restrict__ mean, const float* __restrict__ var,
                               const float* __restrict__ g, const float* __restrict__ b, long total, int C) {
  long i = (long)blockIdx.x * blockDim.x + threadIdx.x;
  if (i >= total) return;
  int c = (int)(i % C);
  float v = b2f(act[i]);
  v = (v - mean[c]) * rsqrtf(var[c] + 1e-5f) * g[c] + b[c];
  v = v >= 0.f ? v : 0.2f * v;
  act[i] = f2b(v);
}
// LSTM cell epilogue: gates (64 x 4096 fp32, i|f|g|o), add biases, update c,
// fan h out to fp32 store + up to 3 bf16 destinations (recurrent slots).
__global__ void lstm_cell(const float* __restrict__ gates,
                          const float* __restrict__ bih, const float* __restrict__ bhh,
                          float* __restrict__ c, float* __restrict__ hf32,
                          u16* hb1, int ld1, u16* hb2, int ld2, u16* hb3, int ld3) {
  int i = blockIdx.x * blockDim.x + threadIdx.x;   // 64*1024 threads
  if (i >= 64 * 1024) return;
  int b = i >> 10, j = i & 1023;
  size_t base = (size_t)b * 4096;
  float gi = gates[base + j]        + bih[j]        + bhh[j];
  float gf = gates[base + 1024 + j] + bih[1024 + j] + bhh[1024 + j];
  float gg = gates[base + 2048 + j] + bih[2048 + j] + bhh[2048 + j];
  float go = gates[base + 3072 + j] + bih[3072 + j] + bhh[3072 + j];
  size_t ci = (size_t)b * 1024 + j;
  float cn = sigmoidf_(gf) * c[ci] + sigmoidf_(gi) * tanhf(gg);
  float h  = sigmoidf_(go) * tanhf(cn);
  c[ci] = cn;
  if (hf32) hf32[ci] = h;
  u16 hb = f2b(h);
  if (hb1) hb1[(size_t)b * ld1 + j] = hb;
  if (hb2) hb2[(size_t)b * ld2 + j] = hb;
  if (hb3) hb3[(size_t)b * ld3 + j] = hb;
}
// 1024 -> 3 linear head. mode 0: rows indexed (b,t,k) over store[k][t][b][:];
// mode 1: rows indexed (b,k) over store[k][b][:].
__global__ __launch_bounds__(128) void head_kernel(const float* __restrict__ store,
                                                   const float* __restrict__ w, const float* __restrict__ bias,
                                                   float* __restrict__ out, int do_softplus, int mode) {
  int r = blockIdx.x, tid = threadIdx.x;
  const float* vec;
  if (mode == 0) {
    int b = r / 20, rem = r % 20, t = rem >> 1, k = rem & 1;
    vec = store + (((size_t)(k * 10 + t)) * 64 + b) * 1024;
  } else {
    int b = r >> 1, k = r & 1;
    vec = store + ((size_t)k * 64 + b) * 1024;
  }
  float p0 = 0.f, p1 = 0.f, p2 = 0.f;
  for (int j = tid; j < 1024; j += 128) {
    float v = vec[j];
    p0 += v * w[j]; p1 += v * w[1024 + j]; p2 += v * w[2048 + j];
  }
  __shared__ float s0[128], s1[128], s2[128];
  s0[tid] = p0; s1[tid] = p1; s2[tid] = p2; __syncthreads();
  for (int o = 64; o > 0; o >>= 1) {
    if (tid < o) { s0[tid] += s0[tid + o]; s1[tid] += s1[tid + o]; s2[tid] += s2[tid + o]; }
    __syncthreads();
  }
  if (tid == 0) {
    float r0 = s0[0] + bias[0], r1 = s1[0] + bias[1], r2 = s2[0] + bias[2];
    if (do_softplus) { r0 = softplusf_(r0); r1 = softplusf_(r1); r2 = softplusf_(r2); }
    out[(size_t)r * 3 + 0] = r0; out[(size_t)r * 3 + 1] = r1; out[(size_t)r * 3 + 2] = r2;
  }
}

// ---------------------------------------------------------------------------
// Host orchestration
// ---------------------------------------------------------------------------
extern "C" void kernel_launch(void* const* d_in, const int* in_sizes, int n_in,
                              void* d_out, int out_size, void* d_ws, size_t ws_size,
                              hipStream_t stream) {
  (void)in_sizes; (void)n_in; (void)out_size; (void)ws_size;

  const float* x        = (const float*)d_in[0];
  const float* convw[5] = {(const float*)d_in[1], (const float*)d_in[2], (const float*)d_in[3],
                           (const float*)d_in[4], (const float*)d_in[5]};
  const float* bng[3]   = {(const float*)d_in[6], (const float*)d_in[8], (const float*)d_in[10]};
  const float* bnb[3]   = {(const float*)d_in[7], (const float*)d_in[9], (const float*)d_in[11]};
  const float* enc_Wih  = (const float*)d_in[12];
  const float* enc_Whh  = (const float*)d_in[13];
  const float* enc_bih  = (const float*)d_in[14];
  const float* enc_bhh  = (const float*)d_in[15];
  const float* pred_Wih = (const float*)d_in[16];
  const float* pred_Whh = (const float*)d_in[17];
  const float* pred_bih = (const float*)d_in[18];
  const float* pred_bhh = (const float*)d_in[19];
  const float* ip_Wih   = (const float*)d_in[20];
  const float* ip_Whh   = (const float*)d_in[21];
  const float* ip_bih   = (const float*)d_in[22];
  const float* ip_bhh   = (const float*)d_in[23];
  const float* bmu_w  = (const float*)d_in[24]; const float* bmu_b  = (const float*)d_in[25];
  const float* bsig_w = (const float*)d_in[26]; const float* bsig_b = (const float*)d_in[27];
  const float* imu_w  = (const float*)d_in[28]; const float* imu_b  = (const float*)d_in[29];
  const float* isig_w = (const float*)d_in[30]; const float* isig_b = (const float*)d_in[31];
  float* out = (float*)d_out;

  // -- workspace carve-up (256B aligned) --
  char* base = (char*)d_ws;
  size_t off = 0;
  auto alloc = [&](size_t bytes) -> void* {
    off = (off + 255) & ~(size_t)255;
    void* p = base + off;
    off += bytes;
    return p;
  };
  u16*   xbf       = (u16*)alloc(2621440ULL * 2);
  u16*   w0t       = (u16*)alloc(32ULL * 64 * 2);
  u16*   w1t       = (u16*)alloc(1024ULL * 128 * 2);
  u16*   w2t       = (u16*)alloc(2048ULL * 256 * 2);
  u16*   w3t       = (u16*)alloc(4096ULL * 512 * 2);
  u16*   w4t       = (u16*)alloc(8192ULL * 512 * 2);
  u16*   actA      = (u16*)alloc(41943040ULL * 2);
  u16*   actB      = (u16*)alloc(20971520ULL * 2);
  u16*   zbf       = (u16*)alloc(327680ULL * 2);
  float* bn_mean   = (float*)alloc(512 * 4);
  float* bn_var    = (float*)alloc(512 * 4);
  u16*   WencT     = (u16*)alloc(2560ULL * 4096 * 2);
  u16*   WpredT    = (u16*)alloc(3072ULL * 4096 * 2);
  u16*   WipT      = (u16*)alloc(2048ULL * 4096 * 2);
  u16*   Aenc      = (u16*)alloc(64ULL * 2560 * 2);
  u16*   Apred     = (u16*)alloc(64ULL * 3072 * 2);
  u16*   Aip       = (u16*)alloc(64ULL * 2048 * 2);
  float* gates     = (float*)alloc(262144ULL * 4);
  float* c_enc     = (float*)alloc(65536ULL * 4);
  float* c_pred    = (float*)alloc(65536ULL * 4);
  float* c_ip      = (float*)alloc(65536ULL * 4);
  float* hid_c     = (float*)alloc(2ULL * 65536 * 4);
  u16*   prevh_enc = (u16*)alloc(655360ULL * 2);
  u16*   prevh_pred= (u16*)alloc(655360ULL * 2);
  float* enc_store = (float*)alloc(1310720ULL * 4);
  float* pred_store= (float*)alloc(1310720ULL * 4);
  float* ip_store  = (float*)alloc(131072ULL * 4);

  auto blocks = [](long n, int b) { return dim3((unsigned)((n + b - 1) / b)); };

  // -- weight prep (bf16, transposed / stacked) --
  cvt_f2b_flat<<<blocks(2621440, 256), 256, 0, stream>>>(x, xbf, 2621440);
  convw_t<<<blocks(32L * 64, 256),   256, 0, stream>>>(convw[0], w0t, 64, 1, 16, 32);
  convw_t<<<blocks(1024L * 128, 256),256, 0, stream>>>(convw[1], w1t, 128, 64, 1024, 1024);
  convw_t<<<blocks(2048L * 256, 256),256, 0, stream>>>(convw[2], w2t, 256, 128, 2048, 2048);
  convw_t<<<blocks(4096L * 512, 256),256, 0, stream>>>(convw[3], w3t, 512, 256, 4096, 4096);
  convw_t<<<blocks(8192L * 512, 256),256, 0, stream>>>(convw[4], w4t, 512, 512, 8192, 8192);
  linw_t<<<blocks(4096L * 1536, 256), 256, 0, stream>>>(enc_Wih,  WencT,                    4096, 1536, 4096);
  linw_t<<<blocks(4096L * 1024, 256), 256, 0, stream>>>(enc_Whh,  WencT  + 1536ULL * 4096,  4096, 1024, 4096);
  linw_t<<<blocks(4096L * 2048, 256), 256, 0, stream>>>(pred_Wih, WpredT,                   4096, 2048, 4096);
  linw_t<<<blocks(4096L * 1024, 256), 256, 0, stream>>>(pred_Whh, WpredT + 2048ULL * 4096,  4096, 1024, 4096);
  linw_t<<<blocks(4096L * 1024, 256), 256, 0, stream>>>(ip_Wih,   WipT,                     4096, 1024, 4096);
  linw_t<<<blocks(4096L * 1024, 256), 256, 0, stream>>>(ip_Whh,   WipT   + 1024ULL * 4096,  4096, 1024, 4096);

  // -- conv encoder (implicit GEMM, bf16 activations), block tile 128x64 --
  // conv0: 640x1x64x64 -> 640x64x32x32, leaky fused. M=655360
  wmma_conv_bf16<<<dim3(1, 5120), 256, 0, stream>>>(xbf, 64, 64, 1, 32, 32, 2, 1, 16,
                                                    w0t, 64, nullptr, actA, 2, 32);
  // conv1 + BN1 + leaky. M=163840
  wmma_conv_bf16<<<dim3(2, 1280), 256, 0, stream>>>(actA, 32, 32, 64, 16, 16, 2, 1, 1024,
                                                    w1t, 128, nullptr, actB, 1, 1024);
  bn_stats<<<128, 256, 0, stream>>>(actB, 163840, 128, bn_mean, bn_var);
  bn_apply_leaky<<<blocks(163840L * 128, 256), 256, 0, stream>>>(actB, bn_mean, bn_var, bng[0], bnb[0], 163840L * 128, 128);
  // conv2 + BN2 + leaky. M=40960
  wmma_conv_bf16<<<dim3(4, 320), 256, 0, stream>>>(actB, 16, 16, 128, 8, 8, 2, 1, 2048,
                                                   w2t, 256, nullptr, actA, 1, 2048);
  bn_stats<<<256, 256, 0, stream>>>(actA, 40960, 256, bn_mean, bn_var);
  bn_apply_leaky<<<blocks(40960L * 256, 256), 256, 0, stream>>>(actA, bn_mean, bn_var, bng[1], bnb[1], 40960L * 256, 256);
  // conv3 + BN3 + leaky. M=10240
  wmma_conv_bf16<<<dim3(8, 80), 256, 0, stream>>>(actA, 8, 8, 256, 4, 4, 2, 1, 4096,
                                                  w3t, 512, nullptr, actB, 1, 4096);
  bn_stats<<<512, 256, 0, stream>>>(actB, 10240, 512, bn_mean, bn_var);
  bn_apply_leaky<<<blocks(10240L * 512, 256), 256, 0, stream>>>(actB, bn_mean, bn_var, bng[2], bnb[2], 10240L * 512, 512);
  // conv4 -> latent z (640 x 512), raw bf16. M=640
  wmma_conv_bf16<<<dim3(8, 5), 256, 0, stream>>>(actB, 4, 4, 512, 1, 1, 1, 0, 8192,
                                                 w4t, 512, nullptr, zbf, 1, 8192);

  // -- encoder LSTM: K=2 passes over T=10 steps (gate GEMM grid: N=4096/128) --
  zero_u32_flat<<<blocks(327680, 256), 256, 0, stream>>>((u32*)prevh_enc, 327680);
  for (int k = 0; k < 2; ++k) {
    zero_u32_flat<<<blocks(65536, 256), 256, 0, stream>>>((u32*)c_enc, 65536);
    zero_u32_flat<<<blocks(81920, 256), 256, 0, stream>>>((u32*)Aenc, 81920);
    for (int t = 0; t < 10; ++t) {
      // A = [ repr_t | prev_pass_h_t | h_rec ]
      copy_b2b<<<blocks(64L * 512, 256), 256, 0, stream>>>(zbf + (size_t)t * 512, 5120, Aenc, 2560, 64, 512);
      copy_b2b<<<blocks(64L * 1024, 256), 256, 0, stream>>>(prevh_enc + (size_t)t * 65536, 1024, Aenc + 512, 2560, 64, 1024);
      wmma_gemm_bf16<<<dim3(32, 1), 256, 0, stream>>>(Aenc, 2560, WencT, 4096, gates, nullptr, 4096, 2560, 0);
      lstm_cell<<<blocks(65536, 256), 256, 0, stream>>>(gates, enc_bih, enc_bhh, c_enc,
          enc_store + ((size_t)(k * 10 + t)) * 65536,
          Aenc + 1536, 2560,
          prevh_enc + (size_t)t * 65536, 1024,
          nullptr, 0);
    }
    copy_f2f<<<blocks(65536, 256), 256, 0, stream>>>(c_enc, 1024, hid_c + (size_t)k * 65536, 1024, 64, 1024);
  }

  // heads on encoder outputs -> ib_mu, ib_sig
  head_kernel<<<1280, 128, 0, stream>>>(enc_store, bmu_w,  bmu_b,  out + 0,    0, 0);
  head_kernel<<<1280, 128, 0, stream>>>(enc_store, bsig_w, bsig_b, out + 3840, 1, 0);

  // -- ip LSTM over first_h (K=2 steps, carried state) --
  zero_u32_flat<<<blocks(65536, 256), 256, 0, stream>>>((u32*)c_ip, 65536);
  zero_u32_flat<<<blocks(65536, 256), 256, 0, stream>>>((u32*)Aip, 65536);
  for (int k = 0; k < 2; ++k) {
    copy_f2b<<<blocks(64L * 1024, 256), 256, 0, stream>>>(enc_store + ((size_t)(k * 10)) * 65536, 1024, Aip, 2048, 64, 1024);
    wmma_gemm_bf16<<<dim3(32, 1), 256, 0, stream>>>(Aip, 2048, WipT, 4096, gates, nullptr, 4096, 2048, 0);
    lstm_cell<<<blocks(65536, 256), 256, 0, stream>>>(gates, ip_bih, ip_bhh, c_ip,
        ip_store + (size_t)k * 65536,
        Aip + 1024, 2048, nullptr, 0, nullptr, 0);
  }
  head_kernel<<<128, 128, 0, stream>>>(ip_store, imu_w,  imu_b,  out + 15360, 0, 1);
  head_kernel<<<128, 128, 0, stream>>>(ip_store, isig_w, isig_b, out + 15744, 1, 1);

  // -- pred LSTM: K=2 passes over T_OUT=10, state seeded from enc hiddens --
  zero_u32_flat<<<blocks(327680, 256), 256, 0, stream>>>((u32*)prevh_pred, 327680);
  for (int k = 0; k < 2; ++k) {
    zero_u32_flat<<<blocks(98304, 256), 256, 0, stream>>>((u32*)Apred, 98304);
    copy_f2f<<<blocks(65536, 256), 256, 0, stream>>>(hid_c + (size_t)k * 65536, 1024, c_pred, 1024, 64, 1024);
    // po0 = h_rec0 = enc outs at t=9 of pass k
    copy_f2b<<<blocks(64L * 1024, 256), 256, 0, stream>>>(enc_store + ((size_t)(k * 10 + 9)) * 65536, 1024, Apred, 3072, 64, 1024);
    copy_f2b<<<blocks(64L * 1024, 256), 256, 0, stream>>>(enc_store + ((size_t)(k * 10 + 9)) * 65536, 1024, Apred + 2048, 3072, 64, 1024);
    for (int t = 0; t < 10; ++t) {
      copy_b2b<<<blocks(64L * 1024, 256), 256, 0, stream>>>(prevh_pred + (size_t)t * 65536, 1024, Apred + 1024, 3072, 64, 1024);
      wmma_gemm_bf16<<<dim3(32, 1), 256, 0, stream>>>(Apred, 3072, WpredT, 4096, gates, nullptr, 4096, 3072, 0);
      lstm_cell<<<blocks(65536, 256), 256, 0, stream>>>(gates, pred_bih, pred_bhh, c_pred,
          pred_store + ((size_t)(k * 10 + t)) * 65536,
          Apred, 3072,                       // po <- h
          Apred + 2048, 3072,                // h_rec <- h
          prevh_pred + (size_t)t * 65536, 1024);
    }
  }
  head_kernel<<<1280, 128, 0, stream>>>(pred_store, bmu_w,  bmu_b,  out + 7680,  0, 0);
  head_kernel<<<1280, 128, 0, stream>>>(pred_store, bsig_w, bsig_b, out + 11520, 1, 0);
}
